// W4A16Linear_15023795602199
// MI455X (gfx1250) — matrix operand order repaired and checked by reading, not compiled, not run on hardware
//
#include <hip/hip_runtime.h>
#include <hip/hip_bf16.h>

typedef __attribute__((ext_vector_type(16))) _Float16 v16h;
typedef __attribute__((ext_vector_type(8)))  float    v8f;
typedef __attribute__((ext_vector_type(2)))  _Float16 h2;

#define N_OUT 28672
#define K_IN  8192
#define NB    256          // K quant blocks (K/32)
#define KSPLIT 8           // K-split factor for occupancy
#define BPW   (NB / KSPLIT) // 32 blocks per workgroup slice

// Unpack 16 int4 nibbles (2 dwords = 16 nibbles each half? no: 2 dwords = 16
// nibbles) into 16 fp16 values (n-8), using the 0x6400 exponent-bias trick:
// (0x6400 | n) is fp16 (1024+n); packed subtract of 1032 gives (n-8) exactly.
// Pair j of word w packs nibbles (j, j+4) -> halves (2j, 2j+1); the matching
// K-permutation is applied to the A fragment so the dot product is unchanged.
__device__ __forceinline__ v16h dequant16(int wlo, int whi) {
  union { v16h v; h2 p[8]; } r;
  const h2 off = {(_Float16)1032.0f, (_Float16)1032.0f};
#pragma unroll
  for (int j = 0; j < 4; ++j) {
    unsigned pl = (((unsigned)wlo >> (4 * j)) & 0x000F000Fu) | 0x64006400u;
    unsigned ph = (((unsigned)whi >> (4 * j)) & 0x000F000Fu) | 0x64006400u;
    r.p[j]     = __builtin_bit_cast(h2, pl) - off;
    r.p[4 + j] = __builtin_bit_cast(h2, ph) - off;
  }
  return r.v;
}

// Main kernel: each wave handles 32 consecutive outputs over BPW quant blocks.
// One v_wmma_f32_16x16x32_f16 per (block, 16-output tile); K=32 == quant block,
// so the per-block scale is applied as a single scalar FMA on the partial.
__global__ __launch_bounds__(256) void w4a16_gemv_wmma(
    const _Float16* __restrict__ x,
    const int*      __restrict__ wq,      // (NB, N_OUT, 4) int32
    const _Float16* __restrict__ scales,  // (NB, N_OUT) fp16
    float*          __restrict__ part)    // (KSPLIT, N_OUT) fp32 partials
{
  // Permuted activation fragments, shared by all 16 (replicated) A rows:
  // px[b*32 + h*16 + i] = x[32*(b0+b) + 16*h + 8*(i>=8) + perm0(i&7)]
  // perm0 = {0,4,1,5,2,6,3,7} matches the dequant nibble interleave.
  __shared__ __align__(32) _Float16 px[BPW * 32];

  const int tid   = threadIdx.x;
  const int lane  = tid & 31;
  const int wave  = tid >> 5;
  const int b0    = blockIdx.y * BPW;                 // first quant block
  const int o0    = (blockIdx.x * 8 + wave) * 32;     // wave's first output
  const int lmod  = lane & 15;
  const int lhalf = lane >> 4;

  for (int idx = tid; idx < BPW * 32; idx += 256) {
    int b = idx >> 5, r = idx & 31;
    int h = r >> 4, i = r & 15;
    int pi = i & 7;
    int p0 = (pi >> 1) | ((pi & 1) << 2);             // {0,4,1,5,2,6,3,7}
    int k = 32 * (b0 + b) + (i & 8) + (h << 4) + p0;
    px[idx] = x[k];
  }
  __syncthreads();

  float acc = 0.0f;
  const v8f zero = {};
#pragma unroll 2
  for (int b = 0; b < BPW; ++b) {
    const int gb = b0 + b;
    // Weights for (block gb, outputs o0..o0+31): 512 contiguous bytes.
    // Lane picks the 8B piece that lands directly in its B-fragment slot:
    // lanes 0-15 -> words 0-1 (K 0..15) of col lmod; lanes 16-31 -> words 2-3
    // (K 16..31) of col lmod. Each b64 load covers a dense 256B region.
    const int* wbase = wq + (((size_t)gb * N_OUT + o0) << 2);
    const int dof = lmod * 4 + lhalf * 2;
    int2 w01 = *(const int2*)(wbase + dof);           // tile0: cols o0..o0+15
    int2 w23 = *(const int2*)(wbase + 64 + dof);      // tile1: cols o0+16..+31
    float sc = (float)scales[(size_t)gb * N_OUT + o0 + lane];

    v16h a  = *(const v16h*)&px[b * 32 + lhalf * 16]; // LDS broadcast, 32B
    v16h B0 = dequant16(w01.x, w01.y);
    v16h B1 = dequant16(w23.x, w23.y);

    v8f d0 = __builtin_amdgcn_wmma_f32_16x16x32_f16(
        false, a, false, B0, (short)0, zero, false, false);
    v8f d1 = __builtin_amdgcn_wmma_f32_16x16x32_f16(
        false, a, false, B1, (short)0, zero, false, false);

    // All A rows are identical, so d[0] on every lane is y for that lane's
    // column: tile0 col lmod (lanes<16), tile1 col lmod (lanes>=16) = o0+lane.
    float pr = lhalf ? d1[0] : d0[0];
    acc = fmaf(sc, pr, acc);
  }
  part[(size_t)blockIdx.y * N_OUT + o0 + lane] = acc; // each slot written once
}

__global__ __launch_bounds__(256) void w4a16_reduce(
    const float* __restrict__ part,
    const _Float16* __restrict__ bias,
    float* __restrict__ out)
{
  int o = blockIdx.x * 256 + threadIdx.x;
  float s = (float)bias[o];
#pragma unroll
  for (int z = 0; z < KSPLIT; ++z) s += part[(size_t)z * N_OUT + o];
  out[o] = s;
}

extern "C" void kernel_launch(void* const* d_in, const int* in_sizes, int n_in,
                              void* d_out, int out_size, void* d_ws, size_t ws_size,
                              hipStream_t stream) {
  (void)in_sizes; (void)n_in; (void)out_size; (void)ws_size;
  const _Float16* x      = (const _Float16*)d_in[0];
  const int*      wq     = (const int*)d_in[1];
  const _Float16* scales = (const _Float16*)d_in[2];
  const _Float16* bias   = (const _Float16*)d_in[3];
  float* part = (float*)d_ws;        // KSPLIT * N_OUT * 4 B = 896 KB scratch
  float* out  = (float*)d_out;

  dim3 grid(N_OUT / (8 * 32), KSPLIT);   // (112, 8) = 896 workgroups
  w4a16_gemv_wmma<<<grid, 256, 0, stream>>>(x, wq, scales, part);
  w4a16_reduce<<<N_OUT / 256, 256, 0, stream>>>(part, bias, out);
}
